// TernaryLinear_31688268710665
// MI455X (gfx1250) — matrix operand order repaired and checked by reading, not compiled, not run on hardware
//
#include <hip/hip_runtime.h>
#include <hip/hip_bf16.h>

typedef __attribute__((ext_vector_type(16))) _Float16 v16h;
typedef __attribute__((ext_vector_type(8)))  _Float16 v8h;
typedef __attribute__((ext_vector_type(4)))  _Float16 v4h;
typedef __attribute__((ext_vector_type(8)))  float    v8f;

// ---------------- sizes (reference is fixed-shape) ----------------
#define M_DIM 8192
#define K_DIM 4096
#define N_DIM 16384

// ws layout (bytes)
#define SCALE_OFF    0
#define PART_OFF     1024
#define NPART        2048
#define XH_OFF       (1u << 20)                                   // 1 MiB
#define WH_OFF       (XH_OFF + (size_t)M_DIM * K_DIM * 2)         // + 64 MiB

// ---------------- 1) partial sum of |W| ----------------
__global__ __launch_bounds__(256) void absum_partial(const float* __restrict__ w,
                                                     size_t n4, float* __restrict__ partials) {
    __shared__ float red[256];
    const float4* w4 = (const float4*)w;
    size_t i = (size_t)blockIdx.x * blockDim.x + threadIdx.x;
    size_t stride = (size_t)gridDim.x * blockDim.x;
    float s = 0.f;
    for (; i < n4; i += stride) {
        float4 v = w4[i];
        s += fabsf(v.x) + fabsf(v.y) + fabsf(v.z) + fabsf(v.w);
    }
    red[threadIdx.x] = s;
    __syncthreads();
    for (int off = 128; off > 0; off >>= 1) {
        if ((int)threadIdx.x < off) red[threadIdx.x] += red[threadIdx.x + off];
        __syncthreads();
    }
    if (threadIdx.x == 0) partials[blockIdx.x] = red[0];
}

// ---------------- 2) finalize scale ----------------
__global__ __launch_bounds__(256) void finalize_scale(const float* __restrict__ partials,
                                                      float inv_count, float* __restrict__ scale) {
    __shared__ float red[256];
    float s = 0.f;
    for (int i = threadIdx.x; i < NPART; i += 256) s += partials[i];
    red[threadIdx.x] = s;
    __syncthreads();
    for (int off = 128; off > 0; off >>= 1) {
        if ((int)threadIdx.x < off) red[threadIdx.x] += red[threadIdx.x + off];
        __syncthreads();
    }
    if (threadIdx.x == 0) scale[0] = red[0] * inv_count + 1e-8f;
}

// ---------------- 3) ternarize W -> f16 ----------------
__global__ __launch_bounds__(256) void ternarize_to_f16(const float* __restrict__ w,
                                                        _Float16* __restrict__ wh,
                                                        const float* __restrict__ scale_p,
                                                        size_t n4) {
    const float inv_s = 1.0f / scale_p[0];
    const float4* w4 = (const float4*)w;
    size_t i = (size_t)blockIdx.x * blockDim.x + threadIdx.x;
    size_t stride = (size_t)gridDim.x * blockDim.x;
    for (; i < n4; i += stride) {
        float4 v = w4[i];
        v4h pack = { (_Float16)fminf(fmaxf(rintf(v.x * inv_s), -1.f), 1.f),
                     (_Float16)fminf(fmaxf(rintf(v.y * inv_s), -1.f), 1.f),
                     (_Float16)fminf(fmaxf(rintf(v.z * inv_s), -1.f), 1.f),
                     (_Float16)fminf(fmaxf(rintf(v.w * inv_s), -1.f), 1.f) };
        *(v4h*)(wh + i * 4) = pack;
    }
}

// ---------------- 4) convert x -> f16 ----------------
__global__ __launch_bounds__(256) void convert_x_f16(const float* __restrict__ x,
                                                     _Float16* __restrict__ xh, size_t n4) {
    const float4* x4 = (const float4*)x;
    size_t i = (size_t)blockIdx.x * blockDim.x + threadIdx.x;
    size_t stride = (size_t)gridDim.x * blockDim.x;
    for (; i < n4; i += stride) {
        float4 v = x4[i];
        v4h pack = { (_Float16)v.x, (_Float16)v.y, (_Float16)v.z, (_Float16)v.w };
        *(v4h*)(xh + i * 4) = pack;
    }
}

// ---------------- fragment load helpers ----------------
// A (16x32 f16, ISA layout): lane group h=lane/16 holds halves
//   [0..7] = K base+8h .. base+8h+7 ; [8..15] = K base+16+8h ..
// caller passes p = row_ptr + 8h ; loads at p and p+16.
__device__ __forceinline__ v16h load_a_frag(const _Float16* p) {
    v8h lo = *(const v8h*)(p);
    v8h hi = *(const v8h*)(p + 16);
    v16h a;
#pragma unroll
    for (int t = 0; t < 8; ++t) { a[t] = lo[t]; a[t + 8] = hi[t]; }
    return a;
}
// B (32x16 f16, B = W^T): lane holds W row n, same K order as A: [0..7] = K 8h.., [8..15] = K 16+8h..
__device__ __forceinline__ v16h load_b_frag(const _Float16* p) {
    return load_a_frag(p);
}

#define WMMA(ACC, A, B) \
    ACC = __builtin_amdgcn_wmma_f32_16x16x32_f16(false, A, false, B, (short)0, ACC, false, false)

// ---------------- 5) WMMA GEMM: out[M,N] = Xh[M,K] * Wh[N,K]^T ----------------
// block = (32, 8) = 8 waves; block tile 128(M) x 128(N)
// wave tile: 64(M) x 32(N) -> 4 x 2 fragments of 16x16, K step 32,
// software-pipelined with 2 K-step ping-pong buffers.
__global__ __launch_bounds__(256) void gemm_wmma_f16(const _Float16* __restrict__ Xh,
                                                     const _Float16* __restrict__ Wh,
                                                     float* __restrict__ out) {
    const int lane  = threadIdx.x;          // 0..31
    const int wave  = threadIdx.y;          // 0..7
    const int wm    = wave & 1;             // 2 waves along M
    const int wn    = wave >> 1;            // 4 waves along N
    const int m0    = blockIdx.y * 128 + wm * 64;
    const int n0    = blockIdx.x * 128 + wn * 32;
    const int lhalf = lane >> 4;            // 0/1
    const int lmod  = lane & 15;

    v8f acc[4][2] = {};

    const _Float16* arow[4];
#pragma unroll
    for (int i = 0; i < 4; ++i)
        arow[i] = Xh + (size_t)(m0 + 16 * i + lmod) * K_DIM + lhalf * 8;
    const _Float16* brow[2];
#pragma unroll
    for (int j = 0; j < 2; ++j)
        brow[j] = Wh + (size_t)(n0 + 16 * j + lmod) * K_DIM + lhalf * 8;

    v16h a0[4], a1[4], b0[2], b1[2];

    // prologue: K steps 0 and 32
#pragma unroll
    for (int i = 0; i < 4; ++i) { a0[i] = load_a_frag(arow[i]); a1[i] = load_a_frag(arow[i] + 32); }
#pragma unroll
    for (int j = 0; j < 2; ++j) { b0[j] = load_b_frag(brow[j]); b1[j] = load_b_frag(brow[j] + 32); }

    for (int k0 = 64; k0 < K_DIM; k0 += 64) {
        // consume buffer 0, refill with k0
#pragma unroll
        for (int i = 0; i < 4; ++i)
#pragma unroll
            for (int j = 0; j < 2; ++j) WMMA(acc[i][j], a0[i], b0[j]);
#pragma unroll
        for (int i = 0; i < 4; ++i) a0[i] = load_a_frag(arow[i] + k0);
#pragma unroll
        for (int j = 0; j < 2; ++j) b0[j] = load_b_frag(brow[j] + k0);

        // consume buffer 1, refill with k0 + 32
#pragma unroll
        for (int i = 0; i < 4; ++i)
#pragma unroll
            for (int j = 0; j < 2; ++j) WMMA(acc[i][j], a1[i], b1[j]);
#pragma unroll
        for (int i = 0; i < 4; ++i) a1[i] = load_a_frag(arow[i] + k0 + 32);
#pragma unroll
        for (int j = 0; j < 2; ++j) b1[j] = load_b_frag(brow[j] + k0 + 32);
    }

    // epilogue: last two K steps
#pragma unroll
    for (int i = 0; i < 4; ++i)
#pragma unroll
        for (int j = 0; j < 2; ++j) WMMA(acc[i][j], a0[i], b0[j]);
#pragma unroll
    for (int i = 0; i < 4; ++i)
#pragma unroll
        for (int j = 0; j < 2; ++j) WMMA(acc[i][j], a1[i], b1[j]);

    // C/D layout: element r -> row m0+16i+8*lhalf+r, col n0+16j+lmod.
    // Non-temporal: out is streamed once; keep L2 for Xh/Wh working set.
#pragma unroll
    for (int i = 0; i < 4; ++i)
#pragma unroll
        for (int j = 0; j < 2; ++j) {
            const int n = n0 + 16 * j + lmod;
            float* p = out + (size_t)(m0 + 16 * i + 8 * lhalf) * N_DIM + n;
#pragma unroll
            for (int r = 0; r < 8; ++r)
                __builtin_nontemporal_store(acc[i][j][r], p + (size_t)r * N_DIM);
        }
}

extern "C" void kernel_launch(void* const* d_in, const int* in_sizes, int n_in,
                              void* d_out, int out_size, void* d_ws, size_t ws_size,
                              hipStream_t stream) {
    (void)in_sizes; (void)n_in; (void)out_size; (void)ws_size;
    const float* x = (const float*)d_in[0];
    const float* w = (const float*)d_in[1];
    float* out = (float*)d_out;

    char* ws = (char*)d_ws;
    float*    scale    = (float*)(ws + SCALE_OFF);
    float*    partials = (float*)(ws + PART_OFF);
    _Float16* Xh       = (_Float16*)(ws + XH_OFF);
    _Float16* Wh       = (_Float16*)(ws + WH_OFF);

    const size_t NK = (size_t)N_DIM * K_DIM;   // 67,108,864
    const size_t MK = (size_t)M_DIM * K_DIM;   // 33,554,432

    absum_partial<<<NPART, 256, 0, stream>>>(w, NK / 4, partials);
    finalize_scale<<<1, 256, 0, stream>>>(partials, 1.0f / (float)NK, scale);
    ternarize_to_f16<<<4096, 256, 0, stream>>>(w, Wh, scale, NK / 4);
    convert_x_f16<<<4096, 256, 0, stream>>>(x, Xh, MK / 4);

    dim3 grid(N_DIM / 128, M_DIM / 128);   // 128 x 64 blocks
    dim3 block(32, 8);
    gemm_wmma_f16<<<grid, block, 0, stream>>>(Xh, Wh, out);
}
